// WindowAttention_57827439674039
// MI455X (gfx1250) — matrix operand order repaired and checked by reading, not compiled, not run on hardware
//
#include <hip/hip_runtime.h>

// MI455X / gfx1250 fused window-attention.
// One 256-thread workgroup (8 wave32) per window; 320KB dynamic LDS.
// All GEMMs through v_wmma_f32_16x16x32_bf16, weights pre-packed into
// per-lane-contiguous B-fragment layout so every fragment load is 2x b128.
// Stage-2 scatter is split into three wave-uniform loops (Q/K/V) so no
// exec-mask divergence is generated.

typedef __attribute__((ext_vector_type(16))) __bf16 v16bf;
typedef __attribute__((ext_vector_type(8)))  float  v8f;

#define NUM_HEADS 16
#define NTOK      64     // window tokens (8x8)
#define DIM       512
#define HEADDIM   32

// workspace layout (bytes)
#define WQ_OFF    0              // 96*16*32*16 bf16 = 1.5 MB  (W_qkv fragments)
#define WP_OFF    1572864        // 32*16*32*16 bf16 = 0.5 MB  (W_proj fragments)
#define BIAS_OFF  2097152        // 16*64*64 f32     = 256 KB  (expanded rel bias)

#define WQ_ELEMS  (96*16*32*16)
#define WP_ELEMS  (32*16*32*16)
#define BIAS_ELEMS (16*64*64)

static __device__ __forceinline__ unsigned short f2bf(float f) {
    union { float f; unsigned u; } v; v.f = f;
    unsigned u = v.u;
    return (unsigned short)((u + 0x7FFFu + ((u >> 16) & 1u)) >> 16);  // RNE
}

union FragBF { uint4 q[2]; v16bf v; };   // 32 bytes: 16 bf16 per lane
union AccF   { v8f v; float f[8]; };

// ---------------------------------------------------------------------------
// Prep kernel: pack weights to bf16 fragment layout + expand relative bias.
// B-fragment convention (32x16 K x N tile): lane = (n%16) + 16*(k/16),
// half = k%16; stored lane-major, 32B contiguous per lane.
// ---------------------------------------------------------------------------
__global__ __launch_bounds__(256)
void wa_prep_kernel(const float* __restrict__ Wqkv,
                    const float* __restrict__ Wproj,
                    const float* __restrict__ btab,
                    unsigned short* __restrict__ wq,
                    unsigned short* __restrict__ wp,
                    float* __restrict__ bias) {
    int e = blockIdx.x * 256 + threadIdx.x;
    if (e < WQ_ELEMS) {
        int ftile = e >> 13;          // / (16*32*16)
        int rem   = e & 8191;
        int kstep = rem >> 9;
        int rem2  = rem & 511;
        int lane  = rem2 >> 4;
        int h     = rem2 & 15;
        int n     = ftile * 16 + (lane & 15);            // output feature (0..1535)
        int k     = kstep * 32 + ((lane >> 4) << 4) + h; // contraction (0..511)
        wq[e] = f2bf(Wqkv[n * DIM + k]);                 // B[k][n] = W[n][k]
        return;
    }
    e -= WQ_ELEMS;
    if (e < WP_ELEMS) {
        int ftile = e >> 13;
        int rem   = e & 8191;
        int kstep = rem >> 9;
        int rem2  = rem & 511;
        int lane  = rem2 >> 4;
        int h     = rem2 & 15;
        int n     = ftile * 16 + (lane & 15);
        int k     = kstep * 32 + ((lane >> 4) << 4) + h;
        wp[e] = f2bf(Wproj[n * DIM + k]);
        return;
    }
    e -= WP_ELEMS;
    if (e < BIAS_ELEMS) {
        int h   = e >> 12;
        int rem = e & 4095;
        int q   = rem >> 6;
        int k   = rem & 63;
        int qy = q >> 3, qx = q & 7, ky = k >> 3, kx = k & 7;
        int idx = (qy - ky + 7) * 15 + (qx - kx + 7);
        bias[e] = btab[idx * NUM_HEADS + h];
    }
}

// 16 WMMA accumulation for one (ftile, mtile) of a M=64,K=512 GEMM with
// A = bf16 rows in LDS (1KB row stride) and B = pre-packed fragments.
static __device__ __forceinline__ v8f gemm_tile(const char* Abase,
                                                const unsigned short* __restrict__ wfrag,
                                                int ft, int arow, int lane, int g) {
    AccF acc;
    #pragma unroll
    for (int r = 0; r < 8; ++r) acc.f[r] = 0.0f;
    const char* arowp = Abase + arow * 1024;
    const unsigned short* wtile = wfrag + ((ft * 16) << 9) + (lane << 4);
    // prefetch next ftile's first fragment tile into near caches
    __builtin_prefetch(wtile + (16 << 9), 0, 0);
    #pragma unroll 4
    for (int ks = 0; ks < 16; ++ks) {
        FragBF a, bb;
        // A frag: per-lane K segment [16g..16g+15]
        a.q[0] = *(const uint4*)(arowp + (ks * 32 + 16 * g) * 2);
        a.q[1] = *(const uint4*)(arowp + (ks * 32 + 16 * g + 8) * 2);
        const uint4* wb = (const uint4*)(wtile + (ks << 9));
        bb.q[0] = wb[0]; bb.q[1] = wb[1];
        acc.v = __builtin_amdgcn_wmma_f32_16x16x32_bf16(
            false, a.v, false, bb.v, (short)0, acc.v, false, false);
    }
    return acc.v;
}

// ---------------------------------------------------------------------------
// Main fused kernel: one block per window.
// LDS (320KB): XB(x bf16 / attn scratch) | QB(Q) | KB(K frags) | VB(V frags) | OB(attn out)
// ---------------------------------------------------------------------------
__global__ __launch_bounds__(256)
void wa_main_kernel(const float* __restrict__ x,
                    const unsigned short* __restrict__ wq_frag,
                    const unsigned short* __restrict__ wp_frag,
                    const float* __restrict__ bias,
                    const float* __restrict__ b_proj,
                    float* __restrict__ out) {
    extern __shared__ char smem[];
    char* XB = smem;               // 64KB: x bf16 [64][512]; later attn scratch (8KB/wave)
    char* QB = smem + 65536;       // 64KB: Q bf16 [16 head][64 tok][32]
    char* KB = smem + 131072;      // 64KB: K B-frags [head][ntile4][lane32][half16]
    char* VB = smem + 196608;      // 64KB: V B-frags [head][dtile2][kstep2][lane32][half16]
    char* OB = smem + 262144;      // 64KB: attn output bf16 [64][512]

    const int b    = blockIdx.x;
    const int tid  = threadIdx.x;
    const int lane = tid & 31;
    const int wave = tid >> 5;
    const int lm   = lane & 15;
    const int g    = lane >> 4;

    // ---- Stage 1: x fp32 -> bf16 into LDS (coalesced float4) ----
    const float4* x4 = (const float4*)(x + (size_t)b * (NTOK * DIM));
    #pragma unroll
    for (int i = 0; i < 32; ++i) {
        int e = i * 256 + tid;
        float4 f = x4[e];
        uint2 pk;
        pk.x = (unsigned)f2bf(f.x) | ((unsigned)f2bf(f.y) << 16);
        pk.y = (unsigned)f2bf(f.z) | ((unsigned)f2bf(f.w) << 16);
        *(uint2*)(XB + e * 8) = pk;
    }
    __syncthreads();

    // ---- Stage 2: qkv = x @ Wqkv^T (M=64, N=1536, K=512) ----
    // Three wave-uniform passes (Q / K / V) -> no exec divergence in stores.
    const float scale = 0.04419417382415922f;  // 512^-0.5 folded into Q

    // Q: ftiles 0..31 -> natural layout [head][token][d], pre-scaled
    for (int i = 0; i < 4; ++i) {
        const int ft  = wave + 8 * i;          // 0..31 (uniform per wave)
        const int hh  = ft >> 1;               // head  (uniform)
        const int d   = ((ft & 1) << 4) + lm;  // 0..31
        for (int mt = 0; mt < 4; ++mt) {
            AccF acc; acc.v = gemm_tile(XB, wq_frag, ft, mt * 16 + lm, lane, g);
            #pragma unroll
            for (int r = 0; r < 8; ++r) {
                int token = mt * 16 + r + 8 * g;
                *(unsigned short*)(QB + (((hh * 64 + token) * 32 + d) << 1)) =
                    f2bf(acc.f[r] * scale);
            }
        }
    }
    // K: ftiles 32..63 -> dots B-fragment layout [head][ntile][lane'][half=lm]
    for (int i = 0; i < 4; ++i) {
        const int ft  = 32 + wave + 8 * i;
        const int ft2 = ft - 32;               // 0..31
        const int hh  = ft2 >> 1;              // head (uniform)
        const int gg  = ft2 & 1;               // d>>4 (uniform)
        for (int mt = 0; mt < 4; ++mt) {
            AccF acc; acc.v = gemm_tile(XB, wq_frag, ft, mt * 16 + lm, lane, g);
            #pragma unroll
            for (int r = 0; r < 8; ++r) {
                int token = mt * 16 + r + 8 * g;   // key token = B column
                int nt = token >> 4, nl = token & 15;
                *(unsigned short*)(KB + ((((hh * 4 + nt) * 32 + (nl + 16 * gg)) * 16 + lm) << 1)) =
                    f2bf(acc.f[r]);
            }
        }
    }
    // V: ftiles 64..95 -> attn@V B-fragment layout [head][dtile][kstep][lane'][half]
    for (int i = 0; i < 4; ++i) {
        const int ft  = 64 + wave + 8 * i;
        const int ft3 = ft - 64;               // 0..31
        const int hh  = ft3 >> 1;              // head  (uniform)
        const int dt  = ft3 & 1;               // dtile (uniform)
        for (int mt = 0; mt < 4; ++mt) {
            AccF acc; acc.v = gemm_tile(XB, wq_frag, ft, mt * 16 + lm, lane, g);
            #pragma unroll
            for (int r = 0; r < 8; ++r) {
                int token = mt * 16 + r + 8 * g;   // key token = contraction dim
                int ksv = token >> 5, kl = token & 31;
                int gg = kl >> 4, hI = kl & 15;
                *(unsigned short*)(VB + (((((hh * 2 + dt) * 2 + ksv) * 32 + (lm + 16 * gg)) * 16 + hI) << 1)) =
                    f2bf(acc.f[r]);
            }
        }
    }
    __syncthreads();

    // ---- Stage 3: per-head attention (2 heads per wave) ----
    char* ATT = XB + wave * 8192;  // [64 q][64 k] bf16 scratch, per-wave
    for (int hh = wave * 2; hh < wave * 2 + 2; ++hh) {
        // dots + bias + softmax, one q-tile at a time
        for (int qt = 0; qt < 4; ++qt) {
            FragBF a;
            const int qrow = qt * 16 + lm;
            const char* qbase = QB + (hh * 64 + qrow) * 64;
            a.q[0] = *(const uint4*)(qbase + (16 * g) * 2);
            a.q[1] = *(const uint4*)(qbase + (16 * g + 8) * 2);
            AccF acc[4];
            #pragma unroll
            for (int kt = 0; kt < 4; ++kt) {
                FragBF bb;
                const uint4* kb = (const uint4*)(KB + (((hh * 4 + kt) * 32 + lane) << 5));
                bb.q[0] = kb[0]; bb.q[1] = kb[1];
                v8f z = {0.f, 0.f, 0.f, 0.f, 0.f, 0.f, 0.f, 0.f};
                acc[kt].v = __builtin_amdgcn_wmma_f32_16x16x32_bf16(
                    false, a.v, false, bb.v, (short)0, z, false, false);
                #pragma unroll
                for (int r = 0; r < 8; ++r) {
                    int q = qt * 16 + r + 8 * g;
                    int k = kt * 16 + lm;
                    acc[kt].f[r] += bias[hh * 4096 + q * 64 + k];
                }
            }
            // softmax across 64 keys: row lives in 16 lanes (one half-wave group) x 4 tiles
            #pragma unroll
            for (int r = 0; r < 8; ++r) {
                float m = fmaxf(fmaxf(acc[0].f[r], acc[1].f[r]),
                                fmaxf(acc[2].f[r], acc[3].f[r]));
                #pragma unroll
                for (int msk = 1; msk < 16; msk <<= 1)
                    m = fmaxf(m, __shfl_xor(m, msk, 16));
                float e0 = __expf(acc[0].f[r] - m);
                float e1 = __expf(acc[1].f[r] - m);
                float e2 = __expf(acc[2].f[r] - m);
                float e3 = __expf(acc[3].f[r] - m);
                float s = e0 + e1 + e2 + e3;
                #pragma unroll
                for (int msk = 1; msk < 16; msk <<= 1)
                    s += __shfl_xor(s, msk, 16);
                float inv = 1.0f / s;
                int q = qt * 16 + r + 8 * g;
                *(unsigned short*)(ATT + ((q * 64 + 0 * 16 + lm) << 1)) = f2bf(e0 * inv);
                *(unsigned short*)(ATT + ((q * 64 + 1 * 16 + lm) << 1)) = f2bf(e1 * inv);
                *(unsigned short*)(ATT + ((q * 64 + 2 * 16 + lm) << 1)) = f2bf(e2 * inv);
                *(unsigned short*)(ATT + ((q * 64 + 3 * 16 + lm) << 1)) = f2bf(e3 * inv);
            }
        }
        // out_head = attn @ V  (M=64, N=32, K=64)
        for (int qt = 0; qt < 4; ++qt) {
            AccF oacc[2];
            #pragma unroll
            for (int dt = 0; dt < 2; ++dt)
                #pragma unroll
                for (int r = 0; r < 8; ++r) oacc[dt].f[r] = 0.0f;
            const int arow = qt * 16 + lm;
            #pragma unroll
            for (int ks = 0; ks < 2; ++ks) {
                FragBF a;
                a.q[0] = *(const uint4*)(ATT + arow * 128 + (ks * 32 + 16 * g) * 2);
                a.q[1] = *(const uint4*)(ATT + arow * 128 + (ks * 32 + 16 * g + 8) * 2);
                #pragma unroll
                for (int dt = 0; dt < 2; ++dt) {
                    FragBF bb;
                    const uint4* vb = (const uint4*)(VB + ((((hh * 2 + dt) * 2 + ks) * 32 + lane) << 5));
                    bb.q[0] = vb[0]; bb.q[1] = vb[1];
                    oacc[dt].v = __builtin_amdgcn_wmma_f32_16x16x32_bf16(
                        false, a.v, false, bb.v, (short)0, oacc[dt].v, false, false);
                }
            }
            #pragma unroll
            for (int dt = 0; dt < 2; ++dt)
                #pragma unroll
                for (int r = 0; r < 8; ++r) {
                    int token = qt * 16 + r + 8 * g;
                    int c = hh * 32 + dt * 16 + lm;
                    *(unsigned short*)(OB + ((token * 512 + c) << 1)) = f2bf(oacc[dt].f[r]);
                }
        }
    }
    __syncthreads();

    // ---- Stage 4: out = attn_out @ Wproj^T + b_proj (M=64, N=512, K=512) ----
    for (int nt = wave * 4; nt < wave * 4 + 4; ++nt) {
        float bp = b_proj[nt * 16 + lm];
        for (int mt = 0; mt < 4; ++mt) {
            AccF acc; acc.v = gemm_tile(OB, wp_frag, nt, mt * 16 + lm, lane, g);
            #pragma unroll
            for (int r = 0; r < 8; ++r) {
                int token = mt * 16 + r + 8 * g;
                out[((size_t)b * 64 + token) * 512 + nt * 16 + lm] = acc.f[r] + bp;
            }
        }
    }
}

// ---------------------------------------------------------------------------
extern "C" void kernel_launch(void* const* d_in, const int* in_sizes, int n_in,
                              void* d_out, int out_size, void* d_ws, size_t ws_size,
                              hipStream_t stream) {
    const float* x     = (const float*)d_in[0];
    const float* Wqkv  = (const float*)d_in[1];
    const float* Wproj = (const float*)d_in[2];
    const float* bproj = (const float*)d_in[3];
    const float* btab  = (const float*)d_in[4];

    char* ws = (char*)d_ws;
    unsigned short* wq   = (unsigned short*)(ws + WQ_OFF);
    unsigned short* wp   = (unsigned short*)(ws + WP_OFF);
    float*          bias = (float*)(ws + BIAS_OFF);

    // pack weights + expand bias (tiny; L2-resident afterwards)
    const int prep_elems = WQ_ELEMS + WP_ELEMS + BIAS_ELEMS;
    wa_prep_kernel<<<(prep_elems + 255) / 256, 256, 0, stream>>>(
        Wqkv, Wproj, btab, wq, wp, bias);

    // fused attention: 2048 windows, 8 waves each, 320KB LDS per workgroup
    wa_main_kernel<<<2048, 256, 327680, stream>>>(
        x, wq, wp, bias, bproj, (float*)d_out);
}
